// ScaleAwareDeformableAttention_42640435315242
// MI455X (gfx1250) — matrix operand order repaired and checked by reading, not compiled, not run on hardware
//
#include <hip/hip_runtime.h>

typedef __attribute__((ext_vector_type(16))) _Float16 v16h;
typedef __attribute__((ext_vector_type(8)))  _Float16 v8h;
typedef __attribute__((ext_vector_type(4)))  _Float16 v4h;
typedef __attribute__((ext_vector_type(8)))  float    v8f;

#define BATCHV 2
#define L1V    21760
#define L2V    21760
#define NHV    8
#define QDV    256
#define MROWS  (BATCHV * L1V)   // 43520

// ---------------------------------------------------------------------------
// fp32 -> fp16 convert, 4 elements per thread (all sizes are multiples of 4)
// ---------------------------------------------------------------------------
__global__ void cvt_f32_f16(const float* __restrict__ in,
                            _Float16* __restrict__ out, int n4) {
  int i = blockIdx.x * blockDim.x + threadIdx.x;
  if (i >= n4) return;
  const float4 v = reinterpret_cast<const float4*>(in)[i];
  v4h h;
  h[0] = (_Float16)v.x; h[1] = (_Float16)v.y;
  h[2] = (_Float16)v.z; h[3] = (_Float16)v.w;
  reinterpret_cast<v4h*>(out)[i] = h;
}

// ---------------------------------------------------------------------------
// WMMA GEMM: C[M,N] = A[M,256] * W[N,256]^T + bias[N]   (f16 in, f32 out)
// Block = 256 threads = 8 waves. Wave w computes rows [bx*128+16w, +16),
// cols [by*64, +64) with V_WMMA_F32_16X16X32_F16.
//
// Fragment layouts used here (wave32; A and B share one contiguous k order):
//   A 16x32 f16 : lane L holds M=L%16; half i -> K=(L/16)*16+i
//                 => two contiguous 8-half loads
//   B 32x16 f16 : lane L holds N=L%16; half j -> K=(L/16)*16+j
//                 => one contiguous 16-half (32B-aligned) load
//   C/D 16x16 f32: lane L col N=L%16; VGPR r -> M=r+(L/16)*8
// ---------------------------------------------------------------------------
__global__ __launch_bounds__(256)
void gemm_wmma_f16(const _Float16* __restrict__ A,
                   const _Float16* __restrict__ W,
                   const float* __restrict__ bias,
                   float* __restrict__ C,
                   int M, int N) {
  const int lane = threadIdx.x & 31;
  const int wave = threadIdx.x >> 5;
  const int m0 = blockIdx.x * 128 + wave * 16;
  const int n0 = blockIdx.y * 64;
  if (m0 >= M) return;

  const int lo16 = lane & 15;
  const int hi   = lane >> 4;          // 0 or 1

  const _Float16* aRow = A + (size_t)(m0 + lo16) * QDV + hi * 16;
  const _Float16* w0 = W + (size_t)(n0 +  0 + lo16) * QDV + hi * 16;
  const _Float16* w1 = W + (size_t)(n0 + 16 + lo16) * QDV + hi * 16;
  const _Float16* w2 = W + (size_t)(n0 + 32 + lo16) * QDV + hi * 16;
  const _Float16* w3 = W + (size_t)(n0 + 48 + lo16) * QDV + hi * 16;

  v8f acc0 = {}, acc1 = {}, acc2 = {}, acc3 = {};

#pragma unroll
  for (int kb = 0; kb < QDV; kb += 32) {
    const v8h alo = *reinterpret_cast<const v8h*>(aRow + kb);
    const v8h ahi = *reinterpret_cast<const v8h*>(aRow + kb + 8);
    const v16h a = __builtin_shufflevector(alo, ahi,
        0, 1, 2, 3, 4, 5, 6, 7, 8, 9, 10, 11, 12, 13, 14, 15);
    const v16h b0 = *reinterpret_cast<const v16h*>(w0 + kb);
    const v16h b1 = *reinterpret_cast<const v16h*>(w1 + kb);
    const v16h b2 = *reinterpret_cast<const v16h*>(w2 + kb);
    const v16h b3 = *reinterpret_cast<const v16h*>(w3 + kb);
    acc0 = __builtin_amdgcn_wmma_f32_16x16x32_f16(false, a, false, b0, (short)0, acc0, false, false);
    acc1 = __builtin_amdgcn_wmma_f32_16x16x32_f16(false, a, false, b1, (short)0, acc1, false, false);
    acc2 = __builtin_amdgcn_wmma_f32_16x16x32_f16(false, a, false, b2, (short)0, acc2, false, false);
    acc3 = __builtin_amdgcn_wmma_f32_16x16x32_f16(false, a, false, b3, (short)0, acc3, false, false);
  }

  const int mBase = m0 + hi * 8;
  const float bv0 = bias[n0 +  0 + lo16];
  const float bv1 = bias[n0 + 16 + lo16];
  const float bv2 = bias[n0 + 32 + lo16];
  const float bv3 = bias[n0 + 48 + lo16];
#pragma unroll
  for (int r = 0; r < 8; ++r) {
    float* crow = C + (size_t)(mBase + r) * N + n0 + lo16;
    crow[ 0] = acc0[r] + bv0;
    crow[16] = acc1[r] + bv1;
    crow[32] = acc2[r] + bv2;
    crow[48] = acc3[r] + bv3;
  }
}

// ---------------------------------------------------------------------------
// Zero masked value rows (mask is (B,L2) bytes; all-false in this setup but
// honored anyway). One block per row.
// ---------------------------------------------------------------------------
__global__ void apply_value_mask(float* __restrict__ Vp,
                                 const unsigned char* __restrict__ mask) {
  const int r = blockIdx.x;
  if (mask[r] != 0) Vp[(size_t)r * QDV + threadIdx.x] = 0.0f;
}

// ---------------------------------------------------------------------------
// Per-(b,query,head) wave: softmax over 16 logits (in place in d_out attn
// region), compute sampling locations, bilinear-gather 4 corners from the
// L2-resident projected value tensor (lane = feature dim -> 128B coalesced
// gathers), weighted-accumulate, emit f16 row for the output GEMM.
// ---------------------------------------------------------------------------
__global__ __launch_bounds__(256)
void softmax_sample(const float* __restrict__ Vp,    // (B,L2,256) f32
                    const float* __restrict__ OFF,   // (B,L1,512) f32
                    float* __restrict__ ATT,         // (B,L1,128) logits -> softmax in place
                    const float* __restrict__ refw,  // (B,L1,4)
                    const float* __restrict__ vvr,   // (B,2)
                    _Float16* __restrict__ S) {      // (B,L1,256) f16
  const int wid  = (int)((blockIdx.x * 256 + threadIdx.x) >> 5);
  const int lane = threadIdx.x & 31;
  if (wid >= BATCHV * L1V * NHV) return;
  const int h  = wid & 7;
  const int q  = (wid >> 3) % L1V;
  const int bb = wid / (NHV * L1V);
  const size_t rowq = (size_t)bb * L1V + q;

  // ---- softmax over the 16 (scale,point) logits of this head ----
  const float* lg = ATT + rowq * 128 + h * 16;
  float w[16];
  float mx = -3.4e38f;
#pragma unroll
  for (int j = 0; j < 16; ++j) { w[j] = lg[j]; mx = fmaxf(mx, w[j]); }
  float ssum = 0.0f;
#pragma unroll
  for (int j = 0; j < 16; ++j) { w[j] = __expf(w[j] - mx); ssum += w[j]; }
  const float inv = 1.0f / ssum;
#pragma unroll
  for (int j = 0; j < 16; ++j) w[j] *= inv;
  if (lane < 16) ATT[rowq * 128 + h * 16 + lane] = w[lane];

  // ---- sampling locations ----
  const float cx = refw[rowq * 4 + 0], cy = refw[rowq * 4 + 1];
  const float rw = refw[rowq * 4 + 2], rh = refw[rowq * 4 + 3];
  const float vrx = vvr[bb * 2 + 0],   vry = vvr[bb * 2 + 1];
  const float* offp = OFF + rowq * 512 + h * 32;           // (s,p,2)
  const float* vb = Vp + (size_t)bb * L2V * QDV + h * 32 + lane;

  const int HWs[4] = {128, 64, 32, 16};
  const int STs[4] = {0, 16384, 20480, 21504};

  float out = 0.0f;
  for (int s = 0; s < 4; ++s) {
    const int G = HWs[s];
    const float fg = (float)G;
    const float* vlev = vb + (size_t)STs[s] * QDV;
#pragma unroll
    for (int p = 0; p < 4; ++p) {
      const float ox = offp[(s * 4 + p) * 2 + 0];
      const float oy = offp[(s * 4 + p) * 2 + 1];
      const float lx = (cx + ox * 0.125f * rw) * vrx;
      const float ly = (cy + oy * 0.125f * rh) * vry;
      const float x = lx * fg - 0.5f;
      const float y = ly * fg - 0.5f;
      const float x0f = floorf(x), y0f = floorf(y);
      const float wx = x - x0f,  wy = y - y0f;
      const int x0 = (int)x0f, y0 = (int)y0f;
      const float cw00 = (1.0f - wx) * (1.0f - wy);
      const float cw10 = wx * (1.0f - wy);
      const float cw01 = (1.0f - wx) * wy;
      const float cw11 = wx * wy;
      float acc = 0.0f;
      if (x0     >= 0 && x0     < G && y0     >= 0 && y0     < G) acc += cw00 * vlev[((size_t)(y0    ) * G + x0    ) * QDV];
      if (x0 + 1 >= 0 && x0 + 1 < G && y0     >= 0 && y0     < G) acc += cw10 * vlev[((size_t)(y0    ) * G + x0 + 1) * QDV];
      if (x0     >= 0 && x0     < G && y0 + 1 >= 0 && y0 + 1 < G) acc += cw01 * vlev[((size_t)(y0 + 1) * G + x0    ) * QDV];
      if (x0 + 1 >= 0 && x0 + 1 < G && y0 + 1 >= 0 && y0 + 1 < G) acc += cw11 * vlev[((size_t)(y0 + 1) * G + x0 + 1) * QDV];
      out += w[s * 4 + p] * acc;
    }
  }
  S[rowq * QDV + h * 32 + lane] = (_Float16)out;
}

// ---------------------------------------------------------------------------
extern "C" void kernel_launch(void* const* d_in, const int* in_sizes, int n_in,
                              void* d_out, int out_size, void* d_ws, size_t ws_size,
                              hipStream_t stream) {
  (void)in_sizes; (void)n_in; (void)out_size; (void)ws_size;

  const float* query  = (const float*)d_in[0];
  const float* value  = (const float*)d_in[1];
  // d_in[2] v_shape / d_in[4] v_start_index: fixed pyramid, hardcoded above.
  const unsigned char* vmask = (const unsigned char*)d_in[3];
  const float* vvr    = (const float*)d_in[5];
  const float* refw   = (const float*)d_in[6];
  const float* W_off  = (const float*)d_in[7];
  const float* b_off  = (const float*)d_in[8];
  const float* W_attn = (const float*)d_in[9];
  const float* b_attn = (const float*)d_in[10];
  const float* W_v    = (const float*)d_in[11];
  const float* b_v    = (const float*)d_in[12];
  const float* W_o    = (const float*)d_in[13];
  const float* b_o    = (const float*)d_in[14];

  float* out_proj = (float*)d_out;                                 // (B,L1,256)
  float* attn_out = out_proj + (size_t)MROWS * QDV;                // (B,L1,128)

  // ---- workspace carve-out (all chunks 256B-aligned sizes), ~201 MB ----
  char* ws = (char*)d_ws;
  _Float16* qh    = (_Float16*)ws;  ws += (size_t)MROWS * QDV * 2; // 22.3 MB
  _Float16* vh    = (_Float16*)ws;  ws += (size_t)MROWS * QDV * 2; // 22.3 MB
  _Float16* wvh   = (_Float16*)ws;  ws += (size_t)256 * 256 * 2;
  _Float16* woffh = (_Float16*)ws;  ws += (size_t)512 * 256 * 2;
  _Float16* wath  = (_Float16*)ws;  ws += (size_t)128 * 256 * 2;
  _Float16* woh   = (_Float16*)ws;  ws += (size_t)256 * 256 * 2;
  float*    Vp    = (float*)ws;     ws += (size_t)MROWS * QDV * 4; // 44.6 MB (L2-resident)
  float*    OFFm  = (float*)ws;     ws += (size_t)MROWS * 512 * 4; // 89.1 MB
  _Float16* S     = (_Float16*)ws;  ws += (size_t)MROWS * QDV * 2; // 22.3 MB

  // ---- fp16 conversions ----
  {
    int n4;
    n4 = MROWS * QDV / 4; cvt_f32_f16<<<(n4 + 255) / 256, 256, 0, stream>>>(query,  qh,    n4);
    n4 = MROWS * QDV / 4; cvt_f32_f16<<<(n4 + 255) / 256, 256, 0, stream>>>(value,  vh,    n4);
    n4 = 256 * 256 / 4;   cvt_f32_f16<<<(n4 + 255) / 256, 256, 0, stream>>>(W_v,    wvh,   n4);
    n4 = 512 * 256 / 4;   cvt_f32_f16<<<(n4 + 255) / 256, 256, 0, stream>>>(W_off,  woffh, n4);
    n4 = 128 * 256 / 4;   cvt_f32_f16<<<(n4 + 255) / 256, 256, 0, stream>>>(W_attn, wath,  n4);
    n4 = 256 * 256 / 4;   cvt_f32_f16<<<(n4 + 255) / 256, 256, 0, stream>>>(W_o,    woh,   n4);
  }

  const dim3 blk(256);
  const int gx = MROWS / 128;  // 340

  // value projection -> Vp (f32), then mask
  gemm_wmma_f16<<<dim3(gx, 256 / 64), blk, 0, stream>>>(vh, wvh, b_v, Vp, MROWS, 256);
  apply_value_mask<<<MROWS, 256, 0, stream>>>(Vp, vmask);

  // sampling offsets -> OFFm (f32)
  gemm_wmma_f16<<<dim3(gx, 512 / 64), blk, 0, stream>>>(qh, woffh, b_off, OFFm, MROWS, 512);

  // attention logits -> directly into d_out attn region (softmaxed in place next)
  gemm_wmma_f16<<<dim3(gx, 128 / 64), blk, 0, stream>>>(qh, wath, b_attn, attn_out, MROWS, 128);

  // softmax + deformable bilinear sampling -> S (f16)
  softmax_sample<<<MROWS, 256, 0, stream>>>(Vp, OFFm, attn_out, refw, vvr, S);

  // output projection -> d_out main section
  gemm_wmma_f16<<<dim3(gx, 256 / 64), blk, 0, stream>>>(S, woh, b_o, out_proj, MROWS, 256);
}
